// ExaoneMoESparseMoEBlock_26620207301232
// MI455X (gfx1250) — matrix operand order repaired and checked by reading, not compiled, not run on hardware
//
#include <hip/hip_runtime.h>

// ---------------------------------------------------------------------------
// MI455X (gfx1250) MoE block.
// Memory-bound: 1.66 GB of fp32 weights read once -> ~71us floor @ 23.3 TB/s.
// All GEMMs use v_wmma_f32_16x16x32_bf16 (wave32, 16x16 tile per wave),
// weights converted fp32->bf16 in registers (no extra HBM traffic).
// ---------------------------------------------------------------------------

typedef __attribute__((ext_vector_type(16))) __bf16 v16bf;
typedef __attribute__((ext_vector_type(8)))  float  v8f;

namespace cfg {
constexpr int T = 1024, H = 2048, E = 64, KTOP = 8;
constexpr int G = 8, TG = 4;
constexpr int I = 1024, IS = 2048;
constexpr float SCALE = 2.5f;
constexpr int ASSIGN = T * KTOP;   // 8192 (token, expert) assignments
}

#define DEV_INLINE __device__ __forceinline__

struct BfPair { uint4 lo, hi; };

// A fragment (16x32 bf16, MxK). ISA layout: lanes 0-15 hold row M=lane,
// VGPR0..3 = K 0..7 (pairs), VGPR4..7 = K 16..23; lanes 16-31 same rows,
// K 8..15 and 24..31. So per lane: contiguous 8 elems at kbase and kbase+16.
DEV_INLINE v16bf load_a_frag(const __bf16* __restrict__ row, int k0, int lane) {
  const int kb = k0 + ((lane >> 4) << 3);
  BfPair p;
  p.lo = *reinterpret_cast<const uint4*>(row + kb);
  p.hi = *reinterpret_cast<const uint4*>(row + kb + 16);
  return __builtin_bit_cast(v16bf, p);
}

// B fragment (32x16 bf16, KxN) from a row-major fp32 matrix W[k*ldb + n].
// Lanes 0-15: column n, K 0..7,16..23; lanes 16-31: column n, K 8..15,24..31.
// For fixed e, the 16 lanes read 64 consecutive bytes -> coalesced.
DEV_INLINE v16bf load_b_frag_f32(const float* __restrict__ W, long ldb,
                                 int k0, int col_base, int lane) {
  const int col = col_base + (lane & 15);
  const int kb  = k0 + ((lane >> 4) << 3);
  const float* wp = W + (long)kb * ldb + col;
  v16bf b;
#pragma unroll
  for (int e = 0; e < 16; ++e) { b[e] = (__bf16)(*wp); wp += (e == 7) ? 9 * ldb : ldb; }
  return b;
}

DEV_INLINE v8f wmma_bf16(v16bf a, v16bf b, v8f c) {
  // (neg_a, A, neg_b, B, c_mod, C, reuse_a, reuse_b)
  return __builtin_amdgcn_wmma_f32_16x16x32_bf16(false, a, false, b,
                                                 (short)0, c, false, false);
}

// Fast silu/sigmoid via v_rcp_f32 (avoids IEEE div_scale/div_fixup expansion;
// 1-ulp rcp error is negligible vs bf16 storage).
DEV_INLINE float sigmoid_f(float x) {
  return __builtin_amdgcn_rcpf(1.f + __expf(-x));
}
DEV_INLINE float silu_f(float x) { return x * sigmoid_f(x); }

// ---------------------------------------------------------------------------
// K0: fp32 -> bf16 conversion of activations
// ---------------------------------------------------------------------------
__global__ void cvt_bf16_kernel(const float* __restrict__ src,
                                __bf16* __restrict__ dst, int n) {
  int i = blockIdx.x * blockDim.x + threadIdx.x;
  if (i < n) dst[i] = (__bf16)src[i];
}

// ---------------------------------------------------------------------------
// K1: router — logits, sigmoid, biased grouped top-k, histogram
// One block (64 threads = 2 waves) per token; thread e owns expert e.
// ---------------------------------------------------------------------------
__global__ void gating_kernel(const float* __restrict__ x,
                              const float* __restrict__ gw,
                              const float* __restrict__ eb,
                              int* __restrict__ tidx, float* __restrict__ tw,
                              int* __restrict__ counts) {
  using namespace cfg;
  const int t = blockIdx.x;
  const int e = threadIdx.x;
  __shared__ float s_score[E];
  __shared__ float s_sb[E];
  const float* xr = x + (size_t)t * H;
  float acc = 0.f;
#pragma unroll 4
  for (int h = 0; h < H; ++h) acc += xr[h] * gw[(size_t)h * E + e];  // coalesced over e
  const float s = sigmoid_f(acc);
  s_score[e] = s;
  s_sb[e]    = s + eb[e];
  __syncthreads();
  if (e == 0) {
    // group score = sum of top-2 biased scores per group of 8
    float gs[G];
    for (int g = 0; g < G; ++g) {
      float m1 = -1e30f, m2 = -1e30f;
      for (int i = 0; i < E / G; ++i) {
        float v = s_sb[g * (E / G) + i];
        if (v > m1) { m2 = m1; m1 = v; } else if (v > m2) { m2 = v; }
      }
      gs[g] = m1 + m2;
    }
    unsigned gsel = 0;
    for (int r = 0; r < TG; ++r) {          // top-4 groups (lowest index on tie)
      int best = 0; float bv = -1e30f;
      for (int g = 0; g < G; ++g)
        if (!((gsel >> g) & 1u) && gs[g] > bv) { bv = gs[g]; best = g; }
      gsel |= 1u << best;
    }
    unsigned long long taken = 0ull;
    int sel[KTOP]; float wv[KTOP]; float wsum = 0.f;
    for (int r = 0; r < KTOP; ++r) {        // top-8 experts within selected groups
      int best = 0; float bv = -1e30f;
      for (int i = 0; i < E; ++i) {
        if (!((gsel >> (i >> 3)) & 1u)) continue;
        if ((taken >> i) & 1ull) continue;
        if (s_sb[i] > bv) { bv = s_sb[i]; best = i; }
      }
      taken |= 1ull << best;
      sel[r] = best; wv[r] = s_score[best]; wsum += s_score[best];
    }
    const float scl = SCALE * __builtin_amdgcn_rcpf(wsum);  // renormalize, x2.5
    for (int r = 0; r < KTOP; ++r) {
      tidx[t * KTOP + r] = sel[r];
      tw[t * KTOP + r]   = wv[r] * scl;
      atomicAdd(&counts[sel[r]], 1);
    }
  }
}

// ---------------------------------------------------------------------------
// K2: exclusive scan over 64 expert counts (trivially serial)
// ---------------------------------------------------------------------------
__global__ void scan_kernel(const int* __restrict__ counts,
                            int* __restrict__ offs, int* __restrict__ cursor) {
  using namespace cfg;
  if (threadIdx.x == 0) {
    int run = 0;
    for (int e = 0; e < E; ++e) { offs[e] = run; cursor[e] = run; run += counts[e]; }
    offs[E] = run;
  }
}

// ---------------------------------------------------------------------------
// K3: scatter (counting sort) -> contiguous per-expert token/weight lists
// ---------------------------------------------------------------------------
__global__ void scatter_kernel(const int* __restrict__ tidx,
                               const float* __restrict__ tw,
                               int* __restrict__ cursor,
                               int* __restrict__ stok, float* __restrict__ sw) {
  using namespace cfg;
  int i = blockIdx.x * blockDim.x + threadIdx.x;
  if (i >= ASSIGN) return;
  int e = tidx[i];
  int pos = atomicAdd(&cursor[e], 1);
  stok[pos] = i >> 3;          // i / KTOP
  sw[pos]   = tw[i];
}

// ---------------------------------------------------------------------------
// K4: routed gate+up GEMMs fused with SiLU*up.  grid = (E, Mtiles, I/128),
// 8 waves/block each owning one 16x16 N-tile; K = H = 2048 (64 WMMA steps x2).
// ---------------------------------------------------------------------------
__global__ void __launch_bounds__(256, 2)
routed_gateup_kernel(const __bf16* __restrict__ xb, const float* __restrict__ wg,
                     const float* __restrict__ wu, const int* __restrict__ offs,
                     const int* __restrict__ stok, __bf16* __restrict__ act) {
  using namespace cfg;
  const int ex   = blockIdx.x;
  const int base = offs[ex];
  const int cnt  = offs[ex + 1] - base;
  const int m0   = blockIdx.y << 4;
  if (m0 >= cnt) return;                         // uniform exit: EXEC stays all-1s
  const int lane     = threadIdx.x & 31;
  const int wv       = threadIdx.x >> 5;
  const int col_base = (blockIdx.z * 8 + wv) << 4;
  const int r        = lane & 15;
  const int mr       = min(m0 + r, cnt - 1);     // clamp ragged rows, mask at store
  const int tok      = stok[base + mr];
  const __bf16* arow = xb + (size_t)tok * H;
  const float* wge = wg + (size_t)ex * H * I;
  const float* wue = wu + (size_t)ex * H * I;
  v8f g = {0.f,0.f,0.f,0.f,0.f,0.f,0.f,0.f};
  v8f u = {0.f,0.f,0.f,0.f,0.f,0.f,0.f,0.f};
#pragma unroll 2
  for (int k0 = 0; k0 < H; k0 += 32) {
    const v16bf a  = load_a_frag(arow, k0, lane);
    const v16bf bg = load_b_frag_f32(wge, I, k0, col_base, lane);
    const v16bf bu = load_b_frag_f32(wue, I, k0, col_base, lane);
    g = wmma_bf16(a, bg, g);
    u = wmma_bf16(a, bu, u);
  }
  const int col = col_base + r;
  const int rhi = (lane >> 4) << 3;
  // Activation math hoisted out of the predicated region; only stores masked.
  __bf16 av[8];
#pragma unroll
  for (int j = 0; j < 8; ++j) av[j] = (__bf16)(silu_f(g[j]) * u[j]);
#pragma unroll
  for (int j = 0; j < 8; ++j) {
    const int row = rhi + j;
    if (m0 + row < cnt)
      act[(size_t)(base + m0 + row) * I + col] = av[j];
  }
}

// ---------------------------------------------------------------------------
// K5: shared-expert gate+up fused with SiLU*up.  grid = (T/16, IS/128).
// ---------------------------------------------------------------------------
__global__ void __launch_bounds__(256, 2)
shared_gateup_kernel(const __bf16* __restrict__ xb, const float* __restrict__ wsg,
                     const float* __restrict__ wsu, __bf16* __restrict__ acts) {
  using namespace cfg;
  const int m0       = blockIdx.x << 4;
  const int lane     = threadIdx.x & 31;
  const int wv       = threadIdx.x >> 5;
  const int col_base = (blockIdx.y * 8 + wv) << 4;
  const int r        = lane & 15;
  const __bf16* arow = xb + (size_t)(m0 + r) * H;
  v8f g = {0.f,0.f,0.f,0.f,0.f,0.f,0.f,0.f};
  v8f u = {0.f,0.f,0.f,0.f,0.f,0.f,0.f,0.f};
#pragma unroll 2
  for (int k0 = 0; k0 < H; k0 += 32) {
    const v16bf a  = load_a_frag(arow, k0, lane);
    const v16bf bg = load_b_frag_f32(wsg, IS, k0, col_base, lane);
    const v16bf bu = load_b_frag_f32(wsu, IS, k0, col_base, lane);
    g = wmma_bf16(a, bg, g);
    u = wmma_bf16(a, bu, u);
  }
  const int col = col_base + r;
  const int rhi = (lane >> 4) << 3;
#pragma unroll
  for (int j = 0; j < 8; ++j)
    acts[(size_t)(m0 + rhi + j) * IS + col] = (__bf16)(silu_f(g[j]) * u[j]);
}

// ---------------------------------------------------------------------------
// K6: shared-expert down proj; writes out directly (base value of the output).
// grid = (T/16, H/128).
// ---------------------------------------------------------------------------
__global__ void __launch_bounds__(256, 2)
shared_down_kernel(const __bf16* __restrict__ acts, const float* __restrict__ wsd,
                   float* __restrict__ out) {
  using namespace cfg;
  const int m0       = blockIdx.x << 4;
  const int lane     = threadIdx.x & 31;
  const int wv       = threadIdx.x >> 5;
  const int col_base = (blockIdx.y * 8 + wv) << 4;
  const int r        = lane & 15;
  const __bf16* arow = acts + (size_t)(m0 + r) * IS;
  v8f acc = {0.f,0.f,0.f,0.f,0.f,0.f,0.f,0.f};
#pragma unroll 2
  for (int k0 = 0; k0 < IS; k0 += 32) {
    const v16bf a = load_a_frag(arow, k0, lane);
    const v16bf b = load_b_frag_f32(wsd, H, k0, col_base, lane);
    acc = wmma_bf16(a, b, acc);
  }
  const int col = col_base + r;
  const int rhi = (lane >> 4) << 3;
#pragma unroll
  for (int j = 0; j < 8; ++j)
    out[(size_t)(m0 + rhi + j) * H + col] = acc[j];
}

// ---------------------------------------------------------------------------
// K7: routed down proj; scales by routing weight and atomically accumulates
// into out (experts collide on token rows).  grid = (E, Mtiles, H/128).
// ---------------------------------------------------------------------------
__global__ void __launch_bounds__(256, 2)
routed_down_kernel(const __bf16* __restrict__ act, const float* __restrict__ wd,
                   const int* __restrict__ offs, const int* __restrict__ stok,
                   const float* __restrict__ sw, float* __restrict__ out) {
  using namespace cfg;
  const int ex   = blockIdx.x;
  const int base = offs[ex];
  const int cnt  = offs[ex + 1] - base;
  const int m0   = blockIdx.y << 4;
  if (m0 >= cnt) return;
  const int lane     = threadIdx.x & 31;
  const int wv       = threadIdx.x >> 5;
  const int col_base = (blockIdx.z * 8 + wv) << 4;
  const int r        = lane & 15;
  const int mr       = min(m0 + r, cnt - 1);
  const __bf16* arow = act + (size_t)(base + mr) * I;   // contiguous sorted rows
  const float* wde   = wd + (size_t)ex * I * H;
  v8f acc = {0.f,0.f,0.f,0.f,0.f,0.f,0.f,0.f};
#pragma unroll 2
  for (int k0 = 0; k0 < I; k0 += 32) {
    const v16bf a = load_a_frag(arow, k0, lane);
    const v16bf b = load_b_frag_f32(wde, H, k0, col_base, lane);
    acc = wmma_bf16(a, b, acc);
  }
  const int col = col_base + r;
  const int rhi = (lane >> 4) << 3;
#pragma unroll
  for (int j = 0; j < 8; ++j) {
    const int row = rhi + j;
    if (m0 + row < cnt) {
      const int   t = stok[base + m0 + row];
      const float w = sw[base + m0 + row];
      unsafeAtomicAdd(&out[(size_t)t * H + col], acc[j] * w);  // global_atomic_add_f32
    }
  }
}

// ---------------------------------------------------------------------------
// Host launcher
// ---------------------------------------------------------------------------
extern "C" void kernel_launch(void* const* d_in, const int* in_sizes, int n_in,
                              void* d_out, int out_size, void* d_ws, size_t ws_size,
                              hipStream_t stream) {
  using namespace cfg;
  const float* x   = (const float*)d_in[0];
  const float* gw  = (const float*)d_in[1];
  const float* eb  = (const float*)d_in[2];
  const float* wg  = (const float*)d_in[3];
  const float* wu  = (const float*)d_in[4];
  const float* wd  = (const float*)d_in[5];
  const float* wsg = (const float*)d_in[6];
  const float* wsu = (const float*)d_in[7];
  const float* wsd = (const float*)d_in[8];
  float* out = (float*)d_out;

  // Workspace carve (~25.3 MB total)
  char* p = (char*)d_ws;
  auto carve = [&](size_t bytes) {
    char* r = p; p += (bytes + 255) & ~(size_t)255; return (void*)r;
  };
  __bf16* xb     = (__bf16*)carve((size_t)T * H * 2);        // 4 MB
  __bf16* act    = (__bf16*)carve((size_t)ASSIGN * I * 2);   // 16.8 MB
  __bf16* acts   = (__bf16*)carve((size_t)T * IS * 2);       // 4 MB
  int*    tidx   = (int*)  carve((size_t)ASSIGN * 4);
  float*  tw     = (float*)carve((size_t)ASSIGN * 4);
  int*    counts = (int*)  carve(E * 4);
  int*    offs   = (int*)  carve((E + 1) * 4);
  int*    cursor = (int*)  carve(E * 4);
  int*    stok   = (int*)  carve((size_t)ASSIGN * 4);
  float*  sw     = (float*)carve((size_t)ASSIGN * 4);

  hipMemsetAsync(counts, 0, E * sizeof(int), stream);
  cvt_bf16_kernel<<<(T * H + 255) / 256, 256, 0, stream>>>(x, xb, T * H);
  gating_kernel<<<T, 64, 0, stream>>>(x, gw, eb, tidx, tw, counts);
  scan_kernel<<<1, 32, 0, stream>>>(counts, offs, cursor);
  scatter_kernel<<<(ASSIGN + 255) / 256, 256, 0, stream>>>(tidx, tw, cursor, stok, sw);
  routed_gateup_kernel<<<dim3(E, T / 16, I / 128), 256, 0, stream>>>(xb, wg, wu, offs, stok, act);
  shared_gateup_kernel<<<dim3(T / 16, IS / 128), 256, 0, stream>>>(xb, wsg, wsu, acts);
  shared_down_kernel<<<dim3(T / 16, H / 128), 256, 0, stream>>>(acts, wsd, out);
  routed_down_kernel<<<dim3(E, T / 16, H / 128), 256, 0, stream>>>(act, wd, offs, stok, sw, out);
}
